// GAT_22617297780844
// MI455X (gfx1250) — matrix operand order repaired and checked by reading, not compiled, not run on hardware
//
#include <hip/hip_runtime.h>
#include <hip/hip_bf16.h>
#include <math.h>

typedef __attribute__((ext_vector_type(16))) _Float16 v16h;
typedef __attribute__((ext_vector_type(8)))  float    v8f;

#define N_NODES   100000
#define N_REL     500
#define E1_CNT    288000
#define E_TOT     320000
#define ALPHA_LR  0.2f
#define EPS_DIV   1e-16f

// ---- monotonic float<->unsigned key for atomicMax on floats ----
__device__ __forceinline__ unsigned fkey(float f) {
  unsigned b = __float_as_uint(f);
  return (b & 0x80000000u) ? ~b : (b | 0x80000000u);
}
__device__ __forceinline__ float fdec(unsigned k) {
  return __uint_as_float((k & 0x80000000u) ? (k ^ 0x80000000u) : ~k);
}

// =====================================================================
// Fused gather + GEMM:  edge_h[e, 0:128] = [x[src]|x[dst]|emb(e)] @ Wc
// One wave -> 16 edges x 128 cols via v_wmma_f32_16x16x32_f16.
// Wc is f16 column-major [128][384] so B fragments are contiguous 32B.
// =====================================================================
__global__ __launch_bounds__(128) void edge_gemm_kernel(
    const float* __restrict__ xfeat,      // [N,128] node features
    const int*   __restrict__ edge_index, // [2,E]
    const int*   __restrict__ edge_type,  // [E1]
    const int*   __restrict__ nhop,       // [E2,2]
    const float* __restrict__ rmat,       // [N_REL,128] relation embeddings
    const _Float16* __restrict__ Wc,      // [128][384] col-major f16
    float*       __restrict__ edge_h)     // [E,128]
{
  const int lane = threadIdx.x & 31;
  const int wave = threadIdx.x >> 5;
  const int tile = blockIdx.x * 4 + wave;
  if (tile >= E_TOT / 16) return;
  const int e0 = tile * 16;
  const int m  = lane & 15;   // row within tile / column within ntile
  const int hi = lane >> 4;   // half-wave selector
  const int e  = e0 + m;

  const int s = edge_index[e];
  const int d = edge_index[E_TOT + e];
  const float* xs = xfeat + (size_t)s * 128;
  const float* xd = xfeat + (size_t)d * 128;
  const float* rp0;
  const float* rp1 = nullptr;
  if (e < E1_CNT) {
    rp0 = rmat + (size_t)edge_type[e] * 128;
  } else {
    const int k2 = e - E1_CNT;
    rp0 = rmat + (size_t)nhop[2 * k2]     * 128;
    rp1 = rmat + (size_t)nhop[2 * k2 + 1] * 128;
  }

  v8f acc[8] = {};   // 8 n-tiles x 8 VGPRs = 16x128 f32 outputs

  for (int kb = 0; kb < 384; kb += 32) {
    // ---- build A fragment: 16 halves = two 8-float chunks of this row ----
    v16h afrag;
#pragma unroll
    for (int ch = 0; ch < 2; ++ch) {
      const int o = kb + hi * 16 + ch * 8;   // 8-aligned, single region
      const float* p;
      const float* q = nullptr;
      if (o < 128)      { p = xs + o; }
      else if (o < 256) { p = xd + (o - 128); }
      else              { p = rp0 + (o - 256);
                          if (rp1) q = rp1 + (o - 256); }
      float4 f0 = reinterpret_cast<const float4*>(p)[0];
      float4 f1 = reinterpret_cast<const float4*>(p)[1];
      if (q) {
        float4 g0 = reinterpret_cast<const float4*>(q)[0];
        float4 g1 = reinterpret_cast<const float4*>(q)[1];
        f0.x += g0.x; f0.y += g0.y; f0.z += g0.z; f0.w += g0.w;
        f1.x += g1.x; f1.y += g1.y; f1.z += g1.z; f1.w += g1.w;
      }
      afrag[ch * 8 + 0] = (_Float16)f0.x; afrag[ch * 8 + 1] = (_Float16)f0.y;
      afrag[ch * 8 + 2] = (_Float16)f0.z; afrag[ch * 8 + 3] = (_Float16)f0.w;
      afrag[ch * 8 + 4] = (_Float16)f1.x; afrag[ch * 8 + 5] = (_Float16)f1.y;
      afrag[ch * 8 + 6] = (_Float16)f1.z; afrag[ch * 8 + 7] = (_Float16)f1.w;
    }
    // ---- 8 n-tiles of B, each a contiguous 32B slab of col-major Wc ----
    const _Float16* wb = Wc + kb + hi * 16;
#pragma unroll
    for (int nt = 0; nt < 8; ++nt) {
      const int n = nt * 16 + m;
      v16h bfrag = *reinterpret_cast<const v16h*>(wb + (size_t)n * 384);
      acc[nt] = __builtin_amdgcn_wmma_f32_16x16x32_f16(
          false, afrag, false, bfrag, (short)0, acc[nt], false, false);
    }
  }

  // ---- store: C layout lane<16: M=v, lane>=16: M=v+8 ; N = lane&15 ----
#pragma unroll
  for (int nt = 0; nt < 8; ++nt) {
#pragma unroll
    for (int v = 0; v < 8; ++v) {
      const int row = e0 + v + hi * 8;
      edge_h[(size_t)row * 128 + nt * 16 + m] = acc[nt][v];
    }
  }
}

// ---- scores: s[e,h] = leakyrelu(edge_h[e] . a), segment max over src ----
__global__ void score_kernel(const float* __restrict__ edge_h,
                             const int*   __restrict__ edge_index,
                             const float* __restrict__ avec,    // [128]
                             float*       __restrict__ scores,  // [E*H]
                             unsigned*    __restrict__ segmax,  // [N*H]
                             int H, int shift)
{
  const int e = blockIdx.x * blockDim.x + threadIdx.x;
  if (e >= E_TOT) return;
  float acc[2] = {0.f, 0.f};
  const float4* row = reinterpret_cast<const float4*>(edge_h + (size_t)e * 128);
  for (int c4 = 0; c4 < 32; ++c4) {
    const float4 v = row[c4];
    const int c = c4 * 4;
    const int h = c >> shift;
    acc[h] += v.x * avec[c] + v.y * avec[c + 1] + v.z * avec[c + 2] + v.w * avec[c + 3];
  }
  const int s = edge_index[e];
  for (int h = 0; h < H; ++h) {
    float sc = acc[h];
    sc = sc >= 0.f ? sc : ALPHA_LR * sc;
    scores[(size_t)e * H + h] = sc;
    atomicMax(&segmax[(size_t)s * H + h], fkey(sc));
  }
}

// ---- w = exp(s - segmax[src]);  denom[src] += w  ----
__global__ void expsum_kernel(float* __restrict__ scores,
                              const unsigned* __restrict__ segmax,
                              const int* __restrict__ edge_index,
                              float* __restrict__ denom, int H)
{
  const int t = blockIdx.x * blockDim.x + threadIdx.x;
  if (t >= E_TOT * H) return;
  const int e = t / H, h = t - e * H;
  const int s = edge_index[e];
  const float w = expf(scores[t] - fdec(segmax[(size_t)s * H + h]));
  scores[t] = w;
  atomicAdd(&denom[(size_t)s * H + h], w);
}

// ---- out[src, c] += (w / (denom[src] + eps)) * edge_h[e, c] ----
__global__ void attagg_kernel(const float* __restrict__ edge_h,
                              const float* __restrict__ w,
                              const float* __restrict__ denom,
                              const int*   __restrict__ edge_index,
                              float*       __restrict__ out,
                              int H, int shift)
{
  const int t = blockIdx.x * blockDim.x + threadIdx.x;
  if (t >= E_TOT * 128) return;
  const int e = t >> 7;
  const int c = t & 127;
  const int h = c >> shift;
  const int s = edge_index[e];
  const float att = w[(size_t)e * H + h] / (denom[(size_t)s * H + h] + EPS_DIV);
  atomicAdd(&out[(size_t)s * 128 + c], att * edge_h[t]);
}

// ---- in-place ELU ----
__global__ void elu_kernel(float* __restrict__ buf, int n)
{
  const int t = blockIdx.x * blockDim.x + threadIdx.x;
  if (t >= n) return;
  const float x = buf[t];
  buf[t] = x > 0.f ? x : expm1f(x);
}

// ---- pre-transpose weights to f16 column-major [128][384] ----
__global__ void prep_w_kernel(const float* __restrict__ W_heads, // [2,384,64]
                              const float* __restrict__ W_out,   // [384,128]
                              _Float16* __restrict__ Wc1,
                              _Float16* __restrict__ Wc2)
{
  const int t = blockIdx.x * blockDim.x + threadIdx.x;
  if (t >= 2 * 128 * 384) return;
  const int which = t / (128 * 384);
  const int idx = t - which * 128 * 384;
  const int n = idx / 384, k = idx - n * 384;
  if (which == 0) {
    const int h = n >> 6, j = n & 63;
    Wc1[idx] = (_Float16)W_heads[(size_t)h * 384 * 64 + (size_t)k * 64 + j];
  } else {
    Wc2[idx] = (_Float16)W_out[(size_t)k * 128 + n];
  }
}

// ---- r2 = r @ W_r  ([500,128]@[128,128], tiny) ----
__global__ void r2_kernel(const float* __restrict__ r,
                          const float* __restrict__ W_r,
                          float* __restrict__ r2)
{
  const int t = blockIdx.x * blockDim.x + threadIdx.x;
  if (t >= N_REL * 128) return;
  const int i = t >> 7, j = t & 127;
  float acc = 0.f;
  for (int k = 0; k < 128; ++k) acc += r[i * 128 + k] * W_r[k * 128 + j];
  r2[t] = acc;
}

extern "C" void kernel_launch(void* const* d_in, const int* in_sizes, int n_in,
                              void* d_out, int out_size, void* d_ws, size_t ws_size,
                              hipStream_t stream) {
  (void)in_sizes; (void)n_in; (void)out_size; (void)ws_size;
  const int*   edge_index = (const int*)  d_in[0];
  const float* x          = (const float*)d_in[1];
  const float* r          = (const float*)d_in[2];
  const int*   edge_type  = (const int*)  d_in[3];
  const int*   nhop       = (const int*)  d_in[4];
  const float* W_heads    = (const float*)d_in[5];
  const float* a_heads    = (const float*)d_in[6];  // [2,64,1] == flat [128]
  const float* W_out      = (const float*)d_in[7];
  const float* a_out      = (const float*)d_in[8];  // [128]
  const float* W_r        = (const float*)d_in[9];

  float* out = (float*)d_out;                       // [N*128] final output
  float* r2  = out + (size_t)N_NODES * 128;         // [500*128] second output

  char* ws = (char*)d_ws;
  size_t off = 0;
  float*    edge_h = (float*)(ws + off);    off += (size_t)E_TOT * 128 * 4;
  float*    scores = (float*)(ws + off);    off += (size_t)E_TOT * 2 * 4;
  unsigned* segmax = (unsigned*)(ws + off); off += (size_t)N_NODES * 2 * 4;
  float*    denom  = (float*)(ws + off);    off += (size_t)N_NODES * 2 * 4;
  float*    x1     = (float*)(ws + off);    off += (size_t)N_NODES * 128 * 4;
  _Float16* Wc1    = (_Float16*)(ws + off); off += (size_t)128 * 384 * 2;
  _Float16* Wc2    = (_Float16*)(ws + off); off += (size_t)128 * 384 * 2;

  hipMemsetAsync(segmax, 0, (size_t)N_NODES * 2 * 4, stream);
  hipMemsetAsync(denom,  0, (size_t)N_NODES * 2 * 4, stream);
  hipMemsetAsync(x1,     0, (size_t)N_NODES * 128 * 4, stream);
  hipMemsetAsync(out,    0, ((size_t)N_NODES * 128 + (size_t)N_REL * 128) * 4, stream);

  prep_w_kernel<<<(2 * 128 * 384 + 255) / 256, 256, 0, stream>>>(W_heads, W_out, Wc1, Wc2);
  r2_kernel<<<(N_REL * 128 + 255) / 256, 256, 0, stream>>>(r, W_r, r2);

  const int gemm_blocks = (E_TOT / 16 + 3) / 4;   // 4 waves/block, 16 edges/wave

  // ---------------- layer 1 (2 heads fused into 128 cols) ----------------
  edge_gemm_kernel<<<gemm_blocks, 128, 0, stream>>>(x, edge_index, edge_type, nhop, r, Wc1, edge_h);
  score_kernel<<<(E_TOT + 255) / 256, 256, 0, stream>>>(edge_h, edge_index, a_heads, scores, segmax, 2, 6);
  expsum_kernel<<<(E_TOT * 2 + 255) / 256, 256, 0, stream>>>(scores, segmax, edge_index, denom, 2);
  attagg_kernel<<<(E_TOT * 128 + 255) / 256, 256, 0, stream>>>(edge_h, scores, denom, edge_index, x1, 2, 6);
  elu_kernel<<<(N_NODES * 128 + 255) / 256, 256, 0, stream>>>(x1, N_NODES * 128);

  // ---------------- layer 2 (single 128-wide "head") ----------------
  hipMemsetAsync(segmax, 0, (size_t)N_NODES * 4, stream);
  hipMemsetAsync(denom,  0, (size_t)N_NODES * 4, stream);
  edge_gemm_kernel<<<gemm_blocks, 128, 0, stream>>>(x1, edge_index, edge_type, nhop, r2, Wc2, edge_h);
  score_kernel<<<(E_TOT + 255) / 256, 256, 0, stream>>>(edge_h, edge_index, a_out, scores, segmax, 1, 7);
  expsum_kernel<<<(E_TOT + 255) / 256, 256, 0, stream>>>(scores, segmax, edge_index, denom, 1);
  attagg_kernel<<<(E_TOT * 128 + 255) / 256, 256, 0, stream>>>(edge_h, scores, denom, edge_index, out, 1, 7);
  elu_kernel<<<(N_NODES * 128 + 255) / 256, 256, 0, stream>>>(out, N_NODES * 128);
}
